// WindowedSelfAttention_52175262712511
// MI455X (gfx1250) — matrix operand order repaired and checked by reading, not compiled, not run on hardware
//
#include <hip/hip_runtime.h>
#include <hip/hip_bf16.h>

// ---------------------------------------------------------------------------
// Windowed self-attention, CDNA5 (gfx1250) WMMA implementation.
// B=2, T=4096, D=1024, H=16, HD=64, W(window)=512, NC=8.
// All matmuls in f16 via v_wmma_f32_16x16x32_f16 (f32 accumulate),
// softmax in f32. Causal block-skipping in the attention kernel.
// ---------------------------------------------------------------------------

typedef _Float16 half_t;
typedef __attribute__((ext_vector_type(16))) _Float16 v16h;
typedef __attribute__((ext_vector_type(8)))  float    v8f;

__device__ __forceinline__ v8f wmma16(v16h a, v16h b, v8f c) {
  // D = A(16x32) * B(32x16) + C, f16 inputs, f32 accumulate
  return __builtin_amdgcn_wmma_f32_16x16x32_f16(
      /*neg_a=*/false, a, /*neg_b=*/false, b,
      /*c_mod=*/(short)0, c, /*reuse_a=*/false, /*reuse_b=*/false);
}

// A-fragment (16x32, MxK) from row-major f16 [rows, ld].
// Lanes 0-15: row = r0+lane, elems = K[c0..c0+7], K[c0+16..c0+23]
// Lanes 16-31: row = r0+lane-16, elems = K[c0+8..c0+15], K[c0+24..c0+31]
__device__ __forceinline__ v16h load_frag_a(const half_t* __restrict__ src,
                                            int ld, int r0, int c0) {
  const int lane = threadIdx.x & 31;
  const int row  = r0 + (lane & 15);
  const int cb   = c0 + ((lane & 16) ? 8 : 0);
  const half_t* p = src + (size_t)row * ld + cb;
  v16h a;
  reinterpret_cast<float4*>(&a)[0] = *reinterpret_cast<const float4*>(p);
  reinterpret_cast<float4*>(&a)[1] = *reinterpret_cast<const float4*>(p + 16);
  return a;
}

// Same A-fragment pattern, but sourced from f16 data resident in LDS.
__device__ __forceinline__ v16h load_frag_a_ldsh(const half_t* __restrict__ src,
                                                 int ld, int r0, int c0) {
  const int lane = threadIdx.x & 31;
  const int row  = r0 + (lane & 15);
  const int cb   = c0 + ((lane & 16) ? 8 : 0);
  const half_t* p = src + (size_t)row * ld + cb;
  v16h a;
  reinterpret_cast<float4*>(&a)[0] = *reinterpret_cast<const float4*>(p);
  reinterpret_cast<float4*>(&a)[1] = *reinterpret_cast<const float4*>(p + 16);
  return a;
}

// B-fragment (32x16, KxN) from N-major f16 storage [N, ld(=K)] — i.e. source
// row n holds B column n contiguously in K. Lanes 0-15: col n0+lane, K k0..k0+7, k0+16..k0+23.
// Lanes 16-31: col n0+lane-16, K k0+8..k0+15, k0+24..k0+31.
__device__ __forceinline__ v16h load_frag_bt(const half_t* __restrict__ src,
                                             int ld, int n0, int k0) {
  const int lane = threadIdx.x & 31;
  const int col  = n0 + (lane & 15);
  const int kb   = k0 + ((lane & 16) ? 8 : 0);
  const half_t* p = src + (size_t)col * ld + kb;
  v16h b;
  reinterpret_cast<float4*>(&b)[0] = *reinterpret_cast<const float4*>(p);
  reinterpret_cast<float4*>(&b)[1] = *reinterpret_cast<const float4*>(p + 16);
  return b;
}

// ---------------------------------------------------------------------------
// Pre-pass conversions
// ---------------------------------------------------------------------------
__global__ __launch_bounds__(256) void cvt_f32_f16(const float* __restrict__ src,
                                                   half_t* __restrict__ dst, int n) {
  int i = blockIdx.x * 256 + threadIdx.x;
  if (i < n) dst[i] = (half_t)src[i];
}

// W [K=1024, N=1024] row-major f32 -> Wt [N, K] f16
__global__ __launch_bounds__(256) void transpose_f32_f16(const float* __restrict__ w,
                                                         half_t* __restrict__ wt) {
  int i = blockIdx.x * 256 + threadIdx.x;   // over 1024*1024
  int nn = i >> 10, kk = i & 1023;
  wt[i] = (half_t)w[kk * 1024 + nn];        // wt[nn*1024 + kk]
}

// ---------------------------------------------------------------------------
// WMMA GEMM: C[8192,1024] = A[8192,1024](f16 rowmajor) x Bt[1024,1024](f16 Nmajor)
// + bias. mode 0: f16 rowmajor out (scaled). mode 1: V-transpose f16 layout
// [B,H,NC,HD,W]. mode 2: f32 rowmajor out.
// Block: 256 threads = 8 waves, 128x128 tile; wave = 32x64 (2x4 WMMA tiles).
// ---------------------------------------------------------------------------
__global__ __launch_bounds__(256) void gemm_wmma(
    const half_t* __restrict__ A, const half_t* __restrict__ Bt,
    const float* __restrict__ bias,
    half_t* __restrict__ outH, float* __restrict__ outF,
    int mode, float scale) {
  const int wave = threadIdx.x >> 5;
  const int wm = wave >> 1;                 // 0..3
  const int wn = wave & 1;                  // 0..1
  const int m0 = blockIdx.y * 128 + wm * 32;
  const int n0 = blockIdx.x * 128 + wn * 64;
  const int lane = threadIdx.x & 31;

  v8f acc[2][4] = {};

  for (int k0 = 0; k0 < 1024; k0 += 32) {
    // Prefetch the streams two K-steps ahead (global_prefetch_b8).
    if (k0 + 64 < 1024) {
      const int row  = m0 + (lane & 15);
      const int cb   = (k0 + 64) + ((lane & 16) ? 8 : 0);
      __builtin_prefetch(A + (size_t)row * 1024 + cb, 0, 1);
      const int col  = n0 + (lane & 15);
      const int kb   = (k0 + 64) + ((lane & 16) ? 16 : 0);
      __builtin_prefetch(Bt + (size_t)col * 1024 + kb, 0, 1);
    }
    v16h a0 = load_frag_a(A, 1024, m0,      k0);
    v16h a1 = load_frag_a(A, 1024, m0 + 16, k0);
#pragma unroll
    for (int j = 0; j < 4; ++j) {
      v16h b = load_frag_bt(Bt, 1024, n0 + 16 * j, k0);
      acc[0][j] = wmma16(a0, b, acc[0][j]);
      acc[1][j] = wmma16(a1, b, acc[1][j]);
    }
  }

  const int nn = lane & 15;
  const int mb = (lane & 16) ? 8 : 0;
#pragma unroll
  for (int i = 0; i < 2; ++i)
#pragma unroll
    for (int j = 0; j < 4; ++j)
#pragma unroll
      for (int e = 0; e < 8; ++e) {
        int m  = m0 + i * 16 + mb + e;      // 0..8191
        int cc = n0 + j * 16 + nn;          // 0..1023
        float v = (acc[i][j][e] + bias[cc]) * scale;
        if (mode == 0) {
          outH[(size_t)m * 1024 + cc] = (half_t)v;
        } else if (mode == 1) {
          int t = m & 4095, b = m >> 12;
          int ch = t >> 9, tw = t & 511;
          int hh = cc >> 6, dd = cc & 63;
          size_t off = ((((size_t)(b * 16 + hh) * 8 + ch) * 64 + dd) << 9) + tw;
          outH[off] = (half_t)v;
        } else {
          outF[(size_t)m * 1024 + cc] = v;
        }
      }
}

// ---------------------------------------------------------------------------
// Attention: one block per (b, h, chunk, 64-row q-tile). 256 threads = 8 waves.
// Phase 1: S[64x512] = (q*0.125) @ K^T  (WMMA; wave w owns a 64x64 col block;
//          blocks with w > qt are fully causally masked and skipped)
// Phase 2: causal softmax, 4 threads/row, P written as f16 to LDS
// Phase 3: O[64x64] = P @ V  (WMMA; A via ds_load_b128 from f16 LDS,
//          B from V^T in global; K-loop bounded by the causal limit)
// ---------------------------------------------------------------------------
#define SLD 520   // padded row stride (elements) -> 16B aligned, bank-spread

__global__ __launch_bounds__(256) void attn_kernel(
    const half_t* __restrict__ qh, const half_t* __restrict__ kh,
    const half_t* __restrict__ vt, half_t* __restrict__ oh) {
  extern __shared__ char smem[];
  float*  S    = reinterpret_cast<float*>(smem);                 // [64][SLD] f32
  half_t* Ph   = reinterpret_cast<half_t*>(smem + 64 * SLD * 4); // [64][SLD] f16
  float*  redm = reinterpret_cast<float*>(smem + 64 * SLD * 6);  // [256]
  float*  reds = redm + 256;                                     // [256]

  int blk = blockIdx.x;
  const int qt = blk & 7;
  const int c  = (blk >> 3) & 7;
  const int h  = (blk >> 6) & 15;
  const int b  = blk >> 10;
  const int bh = b * 16 + h;

  const int wave = threadIdx.x >> 5;
  const int lane = threadIdx.x & 31;

  const int rowQ = b * 4096 + c * 512 + qt * 64;  // row into qh/oh [8192,1024]
  const int rowK = b * 4096 + c * 512;            // base row into kh
  const int hcol = h * 64;
  const int kmax = (qt + 1) * 64;                 // causal column limit (excl.)

  // ---- Phase 1: scores -> LDS. Wave w owns columns [w*64, w*64+64). ----
  if (wave <= qt) {                               // w > qt: fully masked, skip
    const int nbase = wave * 64;
#pragma unroll 1
    for (int mi = 0; mi < 4; ++mi) {
      v8f acc[4] = {};
#pragma unroll
      for (int k0 = 0; k0 < 64; k0 += 32) {
        v16h a = load_frag_a(qh, 1024, rowQ + mi * 16, hcol + k0);
#pragma unroll
        for (int j = 0; j < 4; ++j) {
          v16h bf = load_frag_bt(kh, 1024, rowK + nbase + j * 16, hcol + k0);
          acc[j] = wmma16(a, bf, acc[j]);
        }
      }
      const int nn = lane & 15;
      const int mb = (lane & 16) ? 8 : 0;
#pragma unroll
      for (int j = 0; j < 4; ++j)
#pragma unroll
        for (int e = 0; e < 8; ++e)
          S[(mi * 16 + mb + e) * SLD + nbase + j * 16 + nn] = acc[j][e];
    }
  }
  __syncthreads();

  // ---- Phase 2: causal softmax, 4 threads per row (quadrants of 128 cols). --
  {
    const int r  = threadIdx.x >> 2;              // 0..63
    const int qd = threadIdx.x & 3;               // 0..3
    const int qidx  = qt * 64 + r;                // q index within chunk
    const int ncols = qidx + 1;                   // valid cols: [0, ncols)
    const int c0 = qd * 128;
    const int c1 = (c0 + 128 < ncols) ? (c0 + 128) : ncols;

    float* row = &S[r * SLD];
    float mx = -1e30f;
    for (int j = c0; j < c1; ++j) mx = fmaxf(mx, row[j]);
    redm[threadIdx.x] = mx;
    __syncthreads();
    mx = fmaxf(fmaxf(redm[r * 4 + 0], redm[r * 4 + 1]),
               fmaxf(redm[r * 4 + 2], redm[r * 4 + 3]));

    float sum = 0.f;
    for (int j = c0; j < c1; ++j) {
      float p = __expf(row[j] - mx);
      row[j] = p;
      sum += p;
    }
    reds[threadIdx.x] = sum;
    __syncthreads();
    sum = (reds[r * 4 + 0] + reds[r * 4 + 1]) +
          (reds[r * 4 + 2] + reds[r * 4 + 3]);
    const float inv = 1.f / sum;

    half_t* prow = &Ph[r * SLD];
    for (int j = c0; j < c1; ++j) prow[j] = (half_t)(row[j] * inv);
    for (int j = (c0 > ncols ? c0 : ncols); j < c0 + 128; ++j)
      prow[j] = (half_t)0.f;                      // masked -> P = 0
  }
  __syncthreads();

  // ---- Phase 3: O = P @ V. Wave w -> rows [mi*16,+16), cols [(w&1)*32,+32). --
  {
    const half_t* vbase = vt + (size_t)(bh * 8 + c) * 64 * 512;  // [HD=64][W=512]
    const int mi  = wave >> 1;
    const int njb = (wave & 1) * 2;
    v8f oacc[2] = {};
#pragma unroll 1
    for (int k0 = 0; k0 < kmax; k0 += 32) {       // causal bound on K
      v16h a = load_frag_a_ldsh(Ph, SLD, mi * 16, k0);
#pragma unroll
      for (int j = 0; j < 2; ++j) {
        v16h bf = load_frag_bt(vbase, 512, (njb + j) * 16, k0);
        oacc[j] = wmma16(a, bf, oacc[j]);
      }
    }
    const int nn = lane & 15;
    const int mb = (lane & 16) ? 8 : 0;
#pragma unroll
    for (int j = 0; j < 2; ++j)
#pragma unroll
      for (int e = 0; e < 8; ++e) {
        int m = mi * 16 + mb + e;                    // row in 64-row q tile
        int d = (njb + j) * 16 + nn;                 // 0..63
        oh[(size_t)(rowQ + m) * 1024 + hcol + d] = (half_t)oacc[j][e];
      }
  }
}

// ---------------------------------------------------------------------------
// Host launcher
// ---------------------------------------------------------------------------
extern "C" void kernel_launch(void* const* d_in, const int* in_sizes, int n_in,
                              void* d_out, int out_size, void* d_ws, size_t ws_size,
                              hipStream_t stream) {
  (void)in_sizes; (void)n_in; (void)out_size; (void)ws_size;

  const float* x  = (const float*)d_in[0];
  const float* Wq = (const float*)d_in[1];
  const float* bq = (const float*)d_in[2];
  const float* Wk = (const float*)d_in[3];
  const float* bk = (const float*)d_in[4];
  const float* Wv = (const float*)d_in[5];
  const float* bv = (const float*)d_in[6];
  const float* Wo = (const float*)d_in[7];
  const float* bo = (const float*)d_in[8];
  float* out = (float*)d_out;

  // Workspace layout (halves). oh aliases xh: x is dead after the QKV GEMMs.
  half_t* xh  = (half_t*)d_ws;                       // 8192*1024
  half_t* wqt = xh  + (size_t)8192 * 1024;           // 1024*1024 each
  half_t* wkt = wqt + (size_t)1024 * 1024;
  half_t* wvt = wkt + (size_t)1024 * 1024;
  half_t* wot = wvt + (size_t)1024 * 1024;
  half_t* qh  = wot + (size_t)1024 * 1024;           // 8192*1024
  half_t* kh  = qh  + (size_t)8192 * 1024;           // 8192*1024
  half_t* vt  = kh  + (size_t)8192 * 1024;           // 8192*1024 ([B,H,NC,HD,W])
  half_t* oh  = xh;                                  // alias

  // 1) x -> f16
  {
    int n = 8192 * 1024;
    cvt_f32_f16<<<n / 256, 256, 0, stream>>>(x, xh, n);
  }
  // 2) weights -> transposed f16 (N-major, so B-fragments are contiguous)
  transpose_f32_f16<<<4096, 256, 0, stream>>>(Wq, wqt);
  transpose_f32_f16<<<4096, 256, 0, stream>>>(Wk, wkt);
  transpose_f32_f16<<<4096, 256, 0, stream>>>(Wv, wvt);
  transpose_f32_f16<<<4096, 256, 0, stream>>>(Wo, wot);

  // 3) Q/K/V projections (Q pre-scaled by 1/sqrt(HD)=0.125)
  dim3 grid(8, 64), blk(256);
  gemm_wmma<<<grid, blk, 0, stream>>>(xh, wqt, bq, qh, nullptr, 0, 0.125f);
  gemm_wmma<<<grid, blk, 0, stream>>>(xh, wkt, bk, kh, nullptr, 0, 1.0f);
  gemm_wmma<<<grid, blk, 0, stream>>>(xh, wvt, bv, vt, nullptr, 1, 1.0f);

  // 4) block-diagonal causal attention: B*H*NC*(W/64) = 2048 blocks
  size_t smem = (size_t)64 * SLD * 6 + 2048;   // S(f32) + P(f16) + reductions
  attn_kernel<<<2048, 256, smem, stream>>>(qh, kh, vt, oh);

  // 5) output projection -> f32
  gemm_wmma<<<grid, blk, 0, stream>>>(oh, wot, bo, nullptr, out, 2, 1.0f);
}
